// CrossAttention_61890478735686
// MI455X (gfx1250) — matrix operand order repaired and checked by reading, not compiled, not run on hardware
//
#include <hip/hip_runtime.h>

typedef __attribute__((ext_vector_type(16))) __bf16 v16bf;
typedef __attribute__((ext_vector_type(8)))  float  v8f;
typedef uint4 uint4a __attribute__((may_alias));

#define UNITS  128
#define CDIM   256      // channels = attention rows/cols
#define DDIM   256      // h*w (contraction dim of projections)
#define QKD    64
#define KLEN   9
#define JP     16       // KLEN padded to one WMMA tile
#define NB     16
#define SCALE  0.0625f  // 256^-0.5

// ---------------- workspace layout (bytes) ----------------
#define WS_QT   0                         // qT  [16][256][256] bf16  (n,c,d)  2MB
#define WS_VT   (WS_QT  + NB*CDIM*DDIM*2) // vT  same                         2MB
#define WS_QWT  (WS_VT  + NB*CDIM*DDIM*2) // qwT [128][64][256] bf16 (u,q,d)  4MB
#define WS_KWT  (WS_QWT + UNITS*QKD*DDIM*2)
#define WS_VWT  (WS_KWT + UNITS*QKD*DDIM*2) // vwT [128][16][256] (u,j,d)     1MB

// ---------------- LDS layout (bf16 elements) ----------------
#define WST_OFF  0                         // weight staging [64][256]   32KB
#define VWST_OFF (WST_OFF  + QKD*DDIM)     // vW weight stage [16][256]   8KB
#define KW_OFF   (VWST_OFF + JP*DDIM)      // kW  [256][64]              32KB
#define QW_OFF   (KW_OFF   + CDIM*QKD)     // qW  [256][64]              32KB
#define VW_OFF   (QW_OFF   + CDIM*QKD)     // vW  [16][256] (j-major)     8KB
#define PS_OFF   (VW_OFF   + JP*CDIM)      // P scratch per wave [32][32] 16KB
#define LDS_ELTS  (PS_OFF + 8*32*32)
#define LDS_BYTES (LDS_ELTS*2)             // 131072 bytes

union FragA { v16bf v; __bf16 h[16]; uint4a q[2]; };

__device__ inline unsigned lds_off_of(const void* p) {
  // LDS aperture: addr[31:0] == byte offset within LDS
  return (unsigned)(unsigned long long)p;
}

__device__ inline void async_g2l_b128(unsigned lds_off, const void* g) {
  asm volatile("global_load_async_to_lds_b128 %0, %1, off"
               :: "v"(lds_off), "v"(g) : "memory");
}
__device__ inline void wait_async0() {
  asm volatile("s_wait_asynccnt 0x0" ::: "memory");
}

// ================= prep kernels: one-time f32 -> bf16 (+transpose/pad) =================
__global__ void cvt_transpose_qv(const float* __restrict__ in, __bf16* __restrict__ outT) {
  // in: (n,d,c) f32 ; outT: (n,c,d) bf16 ; idx ordered n,c,d (d fastest)
  const size_t idx = (size_t)blockIdx.x * blockDim.x + threadIdx.x;
  const int d = (int)(idx & 255);
  const int c = (int)((idx >> 8) & 255);
  const size_t n = idx >> 16;
  outT[idx] = (__bf16)in[(n << 16) + ((size_t)d << 8) + c];
}

__global__ void cvt_transpose_w(const float* __restrict__ in, __bf16* __restrict__ outT) {
  // in: (u,d,q) f32, q=64 ; outT: (u,q,d) bf16 ; idx ordered u,q,d
  const size_t idx = (size_t)blockIdx.x * blockDim.x + threadIdx.x;
  const int d = (int)(idx & 255);
  const size_t uu = idx >> 14;
  const int q = (int)((idx >> 8) & 63);
  outT[idx] = (__bf16)in[(uu << 14) + (size_t)d * QKD + q];
}

__global__ void cvt_pad_vw(const float* __restrict__ in, __bf16* __restrict__ outT) {
  // in: (u,d,9) f32 ; outT: (u,j,d) bf16, j padded to 16 ; idx ordered u,j,d
  const size_t idx = (size_t)blockIdx.x * blockDim.x + threadIdx.x;
  const int d = (int)(idx & 255);
  const int j = (int)((idx >> 8) & 15);
  const size_t uu = idx >> 12;
  outT[idx] = (j < KLEN) ? (__bf16)in[(uu * 256 + d) * KLEN + j] : (__bf16)0.f;
}

// ================= helpers for the main kernel =================
// Two 16x16 dot tiles for column tile ct: A = qW strip (K=64), B = kW^T.
__device__ inline void dot_tiles(const __bf16* qWs, const __bf16* kWs,
                                 int c0, int lane16, int half, int ct, v8f s[2]) {
  #pragma unroll
  for (int k2 = 0; k2 < 2; ++k2) {
    const int k0 = k2 * 32;
    FragA aq[2], bk;
    #pragma unroll
    for (int mt = 0; mt < 2; ++mt) {
      const __bf16* p = qWs + (c0 + mt*16 + lane16) * QKD + k0 + half*8;
      aq[mt].q[0] = *(const uint4a*)p;
      aq[mt].q[1] = *(const uint4a*)(p + 16);
    }
    const __bf16* pb = kWs + (ct*16 + lane16) * QKD + k0 + half*8;
    bk.q[0] = *(const uint4a*)pb;
    bk.q[1] = *(const uint4a*)(pb + 16);
    #pragma unroll
    for (int mt = 0; mt < 2; ++mt)
      s[mt] = __builtin_amdgcn_wmma_f32_16x16x32_bf16(
                  false, aq[mt].v, false, bk.v, (short)0, s[mt], false, false);
  }
}

__global__ __launch_bounds__(256, 1)
void CrossAttention_61890478735686_kernel(const __bf16* __restrict__ qT,
                                          const __bf16* __restrict__ vT,
                                          const __bf16* __restrict__ qwT,
                                          const __bf16* __restrict__ kwT,
                                          const __bf16* __restrict__ vwT,
                                          float* __restrict__ out) {
  extern __shared__ __bf16 smem[];
  __bf16* wSt  = smem + WST_OFF;
  __bf16* vwSt = smem + VWST_OFF;
  __bf16* kWs  = smem + KW_OFF;
  __bf16* qWs  = smem + QW_OFF;
  __bf16* vWs  = smem + VW_OFF;

  const int u      = blockIdx.x;
  const int n      = blockIdx.y;
  const int t      = threadIdx.x;
  const int wave   = t >> 5;
  const int lane   = t & 31;
  const int lane16 = lane & 15;
  const int half   = lane >> 4;
  const int c0     = wave * 32;

  __bf16* pS = smem + PS_OFF + wave * (32*32);

  const __bf16* qTn = qT + (size_t)n * CDIM * DDIM;
  const __bf16* vTn = vT + (size_t)n * CDIM * DDIM;

  const unsigned wst_base  = lds_off_of(wSt);
  const unsigned vwst_base = lds_off_of(vwSt);

  // ---- Stage 0: async-copy this u's key/value weight slabs into LDS ----
  {
    const char* ksrc = (const char*)(kwT + (size_t)u * QKD * DDIM);   // 32KB
    #pragma unroll
    for (int i = 0; i < 8; ++i)
      async_g2l_b128(wst_base + (unsigned)((i*256 + t) * 16), ksrc + (i*256 + t) * 16);
    const char* vsrc = (const char*)(vwT + (size_t)u * JP * DDIM);    // 8KB
    #pragma unroll
    for (int i = 0; i < 2; ++i)
      async_g2l_b128(vwst_base + (unsigned)((i*256 + t) * 16), vsrc + (i*256 + t) * 16);
    wait_async0();
    __syncthreads();
  }

  // ---------------- Phase 1: kW (strip x 64) and vW (strip x 16) ----------------
  {
    v8f accK[2][4] = {};
    v8f accV[2]    = {};
    for (int ks = 0; ks < 8; ++ks) {
      const int d0 = ks * 32;
      FragA a[2];
      #pragma unroll
      for (int mt = 0; mt < 2; ++mt) {
        const __bf16* p = vTn + (size_t)(c0 + mt*16 + lane16) * DDIM + d0 + half*8;
        a[mt].q[0] = *(const uint4a*)p;          // K {0..7}+8h
        a[mt].q[1] = *(const uint4a*)(p + 16);   // K {16..23}+8h
      }
      FragA bk[4], bv;
      #pragma unroll
      for (int nt = 0; nt < 4; ++nt) {
        const __bf16* p = wSt + (nt*16 + lane16) * DDIM + d0 + half*8;
        bk[nt].q[0] = *(const uint4a*)p;
        bk[nt].q[1] = *(const uint4a*)(p + 16);
      }
      {
        const __bf16* p = vwSt + lane16 * DDIM + d0 + half*8;
        bv.q[0] = *(const uint4a*)p;
        bv.q[1] = *(const uint4a*)(p + 16);
      }
      #pragma unroll
      for (int mt = 0; mt < 2; ++mt) {
        #pragma unroll
        for (int nt = 0; nt < 4; ++nt)
          accK[mt][nt] = __builtin_amdgcn_wmma_f32_16x16x32_bf16(
                             false, a[mt].v, false, bk[nt].v, (short)0, accK[mt][nt], false, false);
        accV[mt] = __builtin_amdgcn_wmma_f32_16x16x32_bf16(
                       false, a[mt].v, false, bv.v, (short)0, accV[mt], false, false);
      }
    }
    #pragma unroll
    for (int mt = 0; mt < 2; ++mt)
      #pragma unroll
      for (int r = 0; r < 8; ++r) {
        const int c = c0 + mt*16 + r + half*8;
        #pragma unroll
        for (int nt = 0; nt < 4; ++nt)
          kWs[c * QKD + nt*16 + lane16] = (__bf16)accK[mt][nt][r];
        vWs[lane16 * CDIM + c] = (__bf16)accV[mt][r];
      }
  }

  __syncthreads();   // everyone done reading wSt (kw) before restage

  // ---- Stage 1: async-copy query weight slab into the same LDS buffer ----
  {
    const char* qsrc = (const char*)(qwT + (size_t)u * QKD * DDIM);
    #pragma unroll
    for (int i = 0; i < 8; ++i)
      async_g2l_b128(wst_base + (unsigned)((i*256 + t) * 16), qsrc + (i*256 + t) * 16);
    wait_async0();
    __syncthreads();
  }

  // ---------------- Phase 2: qW (strip x 64) ----------------
  {
    v8f accQ[2][4] = {};
    for (int ks = 0; ks < 8; ++ks) {
      const int d0 = ks * 32;
      FragA a[2];
      #pragma unroll
      for (int mt = 0; mt < 2; ++mt) {
        const __bf16* p = qTn + (size_t)(c0 + mt*16 + lane16) * DDIM + d0 + half*8;
        a[mt].q[0] = *(const uint4a*)p;
        a[mt].q[1] = *(const uint4a*)(p + 16);
      }
      FragA b[4];
      #pragma unroll
      for (int nt = 0; nt < 4; ++nt) {
        const __bf16* p = wSt + (nt*16 + lane16) * DDIM + d0 + half*8;
        b[nt].q[0] = *(const uint4a*)p;
        b[nt].q[1] = *(const uint4a*)(p + 16);
      }
      #pragma unroll
      for (int mt = 0; mt < 2; ++mt)
        #pragma unroll
        for (int nt = 0; nt < 4; ++nt)
          accQ[mt][nt] = __builtin_amdgcn_wmma_f32_16x16x32_bf16(
                             false, a[mt].v, false, b[nt].v, (short)0, accQ[mt][nt], false, false);
    }
    #pragma unroll
    for (int mt = 0; mt < 2; ++mt)
      #pragma unroll
      for (int r = 0; r < 8; ++r) {
        const int c = c0 + mt*16 + r + half*8;
        #pragma unroll
        for (int nt = 0; nt < 4; ++nt)
          qWs[c * QKD + nt*16 + lane16] = (__bf16)accQ[mt][nt][r];
      }
  }

  __syncthreads();   // kWs/vWs/qWs visible to all waves

  // ---------------- Pass 1: row max of dot over all 256 columns ----------------
  float rmax[2][8];
  #pragma unroll
  for (int mt = 0; mt < 2; ++mt)
    #pragma unroll
    for (int r = 0; r < 8; ++r) rmax[mt][r] = -__builtin_inff();

  for (int ct = 0; ct < 16; ++ct) {
    v8f s[2] = {};
    dot_tiles(qWs, kWs, c0, lane16, half, ct, s);
    #pragma unroll
    for (int mt = 0; mt < 2; ++mt)
      #pragma unroll
      for (int r = 0; r < 8; ++r)
        rmax[mt][r] = fmaxf(rmax[mt][r], s[mt][r]);
  }
  #pragma unroll
  for (int mt = 0; mt < 2; ++mt)
    #pragma unroll
    for (int r = 0; r < 8; ++r) {
      float m = rmax[mt][r];
      m = fmaxf(m, __shfl_xor(m, 1, 32));
      m = fmaxf(m, __shfl_xor(m, 2, 32));
      m = fmaxf(m, __shfl_xor(m, 4, 32));
      m = fmaxf(m, __shfl_xor(m, 8, 32));
      rmax[mt][r] = m;
    }

  // ---------------- Pass 2: exp, row-sum, and P x vW via WMMA ----------------
  float rsum[2][8] = {};
  v8f oacc[2] = {};
  for (int ct2 = 0; ct2 < 8; ++ct2) {      // 32 attention columns per step (K=32)
    #pragma unroll
    for (int sub = 0; sub < 2; ++sub) {
      v8f s[2] = {};
      dot_tiles(qWs, kWs, c0, lane16, half, ct2*2 + sub, s);
      #pragma unroll
      for (int mt = 0; mt < 2; ++mt)
        #pragma unroll
        for (int r = 0; r < 8; ++r) {
          const float p = __expf(SCALE * (s[mt][r] - rmax[mt][r]));
          rsum[mt][r] += p;
          pS[(mt*16 + r + half*8) * 32 + sub*16 + lane16] = (__bf16)p;
        }
    }
    FragA ap[2], bv;
    #pragma unroll
    for (int mt = 0; mt < 2; ++mt) {
      const __bf16* p = pS + (mt*16 + lane16) * 32 + half*8;
      ap[mt].q[0] = *(const uint4a*)p;
      ap[mt].q[1] = *(const uint4a*)(p + 16);
    }
    const __bf16* pb = vWs + lane16 * CDIM + ct2*32 + half*8;
    bv.q[0] = *(const uint4a*)pb;
    bv.q[1] = *(const uint4a*)(pb + 16);
    #pragma unroll
    for (int mt = 0; mt < 2; ++mt)
      oacc[mt] = __builtin_amdgcn_wmma_f32_16x16x32_bf16(
                     false, ap[mt].v, false, bv.v, (short)0, oacc[mt], false, false);
  }
  #pragma unroll
  for (int mt = 0; mt < 2; ++mt)
    #pragma unroll
    for (int r = 0; r < 8; ++r) {
      float s2 = rsum[mt][r];
      s2 += __shfl_xor(s2, 1, 32);
      s2 += __shfl_xor(s2, 2, 32);
      s2 += __shfl_xor(s2, 4, 32);
      s2 += __shfl_xor(s2, 8, 32);
      rsum[mt][r] = s2;
    }

  // ---------------- Store: out[n, kh, kw, c, u] ----------------
  if (lane16 < KLEN) {
    const int kh = lane16 / 3, kw = lane16 % 3;
    float* ob = out + ((((size_t)n*3 + kh)*3 + kw) * CDIM) * UNITS + u;
    #pragma unroll
    for (int mt = 0; mt < 2; ++mt)
      #pragma unroll
      for (int r = 0; r < 8; ++r) {
        const int c = c0 + mt*16 + r + half*8;
        ob[(size_t)c * UNITS] = oacc[mt][r] / rsum[mt][r];
      }
  }
}

extern "C" void kernel_launch(void* const* d_in, const int* in_sizes, int n_in,
                              void* d_out, int out_size, void* d_ws, size_t ws_size,
                              hipStream_t stream) {
  (void)in_sizes; (void)n_in; (void)ws_size; (void)out_size;
  const float* query   = (const float*)d_in[0];
  const float* value   = (const float*)d_in[1];
  const float* query_w = (const float*)d_in[2];
  const float* key_w   = (const float*)d_in[3];
  const float* value_w = (const float*)d_in[4];
  float* out = (float*)d_out;

  char* ws = (char*)d_ws;
  __bf16* qT  = (__bf16*)(ws + WS_QT);
  __bf16* vT  = (__bf16*)(ws + WS_VT);
  __bf16* qwT = (__bf16*)(ws + WS_QWT);
  __bf16* kwT = (__bf16*)(ws + WS_KWT);
  __bf16* vwT = (__bf16*)(ws + WS_VWT);

  // one-time conversions/transposes (amortized 16-128x by the main kernel)
  cvt_transpose_qv<<<(NB*CDIM*DDIM)/256, 256, 0, stream>>>(query, qT);
  cvt_transpose_qv<<<(NB*CDIM*DDIM)/256, 256, 0, stream>>>(value, vT);
  cvt_transpose_w <<<(UNITS*QKD*DDIM)/256, 256, 0, stream>>>(query_w, qwT);
  cvt_transpose_w <<<(UNITS*QKD*DDIM)/256, 256, 0, stream>>>(key_w,  kwT);
  cvt_pad_vw      <<<(UNITS*JP*DDIM)/256, 256, 0, stream>>>(value_w, vwT);

  hipFuncSetAttribute((const void*)CrossAttention_61890478735686_kernel,
                      hipFuncAttributeMaxDynamicSharedMemorySize, LDS_BYTES);
  dim3 grid(UNITS, NB);   // one block per (u, n)
  CrossAttention_61890478735686_kernel<<<grid, 256, LDS_BYTES, stream>>>(
      qT, vT, qwT, kwT, vwT, out);
}
